// scoring_model_33543694582408
// MI455X (gfx1250) — matrix operand order repaired and checked by reading, not compiled, not run on hardware
//
#include <hip/hip_runtime.h>
#include <math.h>

#define EPSN 1e-8f

typedef __attribute__((ext_vector_type(16))) __bf16 v16bf;
typedef __attribute__((ext_vector_type(8)))  float  v8f;
typedef float f32x4u __attribute__((ext_vector_type(4), aligned(4)));  // 4B-aligned vector load

struct Pack32 { uint4 lo, hi; };

__device__ __forceinline__ unsigned short f2bf(float f) {
  unsigned int u = __float_as_uint(f);
  u += 0x7FFFu + ((u >> 16) & 1u);            // round-to-nearest-even
  return (unsigned short)(u >> 16);
}

__device__ __forceinline__ unsigned int pack2(float a, float b) {
  return (unsigned int)f2bf(a) | ((unsigned int)f2bf(b) << 16);
}

__device__ __forceinline__ v8f wmma_bf16(uint4 alo, uint4 ahi, uint4 blo, uint4 bhi, v8f c) {
  Pack32 pa{alo, ahi}, pb{blo, bhi};
  v16bf a = __builtin_bit_cast(v16bf, pa);
  v16bf b = __builtin_bit_cast(v16bf, pb);
  return __builtin_amdgcn_wmma_f32_16x16x32_bf16(false, a, false, b, (short)0, c, false, false);
}

__device__ __forceinline__ float sigmoidf(float x) { return 1.f / (1.f + expf(-x)); }

// ---------------------------------------------------------------- utility
__global__ void k_zero(float* __restrict__ p, int n) {
  int i = blockIdx.x * blockDim.x + threadIdx.x;
  if (i < n) p[i] = 0.f;
}

// src: f32 [K,N] row-major -> dst: bf16 B^T [N,KP] (zero-padded K)
__global__ void k_convert_t(const float* __restrict__ src, unsigned short* __restrict__ dst,
                            int K, int N, int KP) {
  int i = blockIdx.x * blockDim.x + threadIdx.x;
  if (i >= N * KP) return;
  int n = i / KP, k = i % KP;
  dst[i] = (k < K) ? f2bf(src[(size_t)k * N + n]) : (unsigned short)0;
}

// ---------------------------------------------------------------- s = x_s @ lin_w + lin_b
#define LIN_KP    1440
#define LIN_TILES 45
__global__ void __launch_bounds__(256)
k_gemm_lin(const float* __restrict__ X, const unsigned short* __restrict__ BT,
           const float* __restrict__ bias, float* __restrict__ S, int M, int K)
{
  __shared__ unsigned short Asm[2][128 * 40];  // double-buffered, stride 40 -> conflict-free
  int tid = threadIdx.x;
  int wave = tid >> 5, lane = tid & 31;
  int m0 = blockIdx.x * 128;

  // staging coords: thread covers row tid/2, 16 consecutive k
  int srow = tid >> 1;
  int skk  = (tid & 1) << 4;
  int grow = m0 + srow;
  const float* xbase = X + (size_t)(grow < M ? grow : M - 1) * K;

  v8f acc[8] = {};
  int aoff = (lane >> 4) << 3;
  int koff = (lane >> 4) << 3;
  int arow = wave * 16 + (lane & 15);

  // prologue: stage tile 0 (never the tail)
  {
    const float* xp = xbase + skk;
    f32x4u f0 = *(const f32x4u*)(xp);
    f32x4u f1 = *(const f32x4u*)(xp + 4);
    f32x4u f2 = *(const f32x4u*)(xp + 8);
    f32x4u f3 = *(const f32x4u*)(xp + 12);
    uint4 p0, p1;
    p0.x = pack2(f0[0], f0[1]); p0.y = pack2(f0[2], f0[3]);
    p0.z = pack2(f1[0], f1[1]); p0.w = pack2(f1[2], f1[3]);
    p1.x = pack2(f2[0], f2[1]); p1.y = pack2(f2[2], f2[3]);
    p1.z = pack2(f3[0], f3[1]); p1.w = pack2(f3[2], f3[3]);
    uint4* ap = (uint4*)(Asm[0] + srow * 40 + skk);
    ap[0] = p0; ap[1] = p1;
  }
  __syncthreads();

  for (int i = 0; i < LIN_TILES; ++i) {
    int kt = i * 32;
    int cur = i & 1;
    bool haveNext = (i + 1 < LIN_TILES);
    bool nextTail = (i + 1 == LIN_TILES - 1);
    f32x4u fn[4];
    if (haveNext) {                       // issue next tile's global loads early
      if (nextTail) {                     // branchless clamped loads for ragged K
        int ktn = kt + 32;
        #pragma unroll
        for (int q = 0; q < 4; ++q)
          #pragma unroll
          for (int j = 0; j < 4; ++j) {
            int kg = ktn + skk + q * 4 + j;
            float vv = xbase[kg < K ? kg : K - 1];
            fn[q][j] = (kg < K) ? vv : 0.f;
          }
      } else {
        const float* xp = xbase + kt + 32 + skk;
        fn[0] = *(const f32x4u*)(xp);
        fn[1] = *(const f32x4u*)(xp + 4);
        fn[2] = *(const f32x4u*)(xp + 8);
        fn[3] = *(const f32x4u*)(xp + 12);
      }
    }
    // compute on current buffer: batch all fragment loads, then 8 WMMAs back-to-back
    const unsigned short* ab = Asm[cur] + arow * 40 + aoff;
    uint4 alo = *(const uint4*)(ab);
    uint4 ahi = *(const uint4*)(ab + 16);
    const unsigned short* bb = BT + (size_t)(lane & 15) * LIN_KP + kt + koff;
    uint4 blo[8], bhi[8];
    #pragma unroll
    for (int t = 0; t < 8; ++t) {
      blo[t] = *(const uint4*)(bb + (size_t)t * 16 * LIN_KP);
      bhi[t] = *(const uint4*)(bb + (size_t)t * 16 * LIN_KP + 16);
    }
    #pragma unroll
    for (int t = 0; t < 8; ++t)
      acc[t] = wmma_bf16(alo, ahi, blo[t], bhi[t], acc[t]);
    if (haveNext) {                       // pack + store next tile to other buffer
      uint4 p0, p1;
      p0.x = pack2(fn[0][0], fn[0][1]); p0.y = pack2(fn[0][2], fn[0][3]);
      p0.z = pack2(fn[1][0], fn[1][1]); p0.w = pack2(fn[1][2], fn[1][3]);
      p1.x = pack2(fn[2][0], fn[2][1]); p1.y = pack2(fn[2][2], fn[2][3]);
      p1.z = pack2(fn[3][0], fn[3][1]); p1.w = pack2(fn[3][2], fn[3][3]);
      uint4* ap = (uint4*)(Asm[1 - cur] + srow * 40 + skk);
      ap[0] = p0; ap[1] = p1;
    }
    __syncthreads();
  }

  int r0 = m0 + wave * 16 + ((lane >> 4) << 3);
  int col = lane & 15;
  #pragma unroll
  for (int t = 0; t < 8; ++t) {
    int c = t * 16 + col;
    float b = bias[c];
    #pragma unroll
    for (int r = 0; r < 8; ++r)
      if (r0 + r < M) S[(size_t)(r0 + r) * 128 + c] = acc[t][r] + b;
  }
}

// ---------------------------------------------------------------- fused edge message kernel
#define A0S 296   // bf16 row stride for A0 (288 + pad 8)
#define A1S 168   // bf16 row stride for A1/A2 (160 + pad 8)
__global__ void __launch_bounds__(128)
k_edge(int Etot,
       const int* __restrict__ eidx, const float* __restrict__ e_s,
       const float* __restrict__ e_v, const float* __restrict__ x_v,
       const float* __restrict__ S,
       const unsigned short* __restrict__ B0T, const unsigned short* __restrict__ B1T,
       const unsigned short* __restrict__ B2T,
       const float* __restrict__ bs0, const float* __restrict__ bs1, const float* __restrict__ bs2,
       const float* __restrict__ wh0, const float* __restrict__ wv0,
       const float* __restrict__ wh1, const float* __restrict__ wv1,
       const float* __restrict__ wh2, const float* __restrict__ wv2,
       float* __restrict__ aggS, float* __restrict__ aggV, float* __restrict__ cnt)
{
  __shared__ unsigned short A[64 * A0S];       // reused for A1/A2 (smaller stride)
  __shared__ float VH0[64][15];
  __shared__ float VH[64][6];
  __shared__ int DSTS[64];

  int tid = threadIdx.x;
  int wave = tid >> 5, lane = tid & 31;
  int e0 = blockIdx.x * 64;

  { // ---- build A0 = [s_src(128) | edge_s(8) | s_dst(128) | |vh0|(5) | 0(19)]
    int le = tid & 63;
    int e  = e0 + le; if (e >= Etot) e = Etot - 1;
    unsigned short* ap = A + le * A0S;
    if (tid < 64) {
      int src = eidx[e];
      const float* sp = S + (size_t)src * 128;
      for (int j = 0; j < 128; j += 8) {
        f32x4u v0 = *(const f32x4u*)(sp + j);
        f32x4u v1 = *(const f32x4u*)(sp + j + 4);
        uint4 p;
        p.x = pack2(v0[0], v0[1]); p.y = pack2(v0[2], v0[3]);
        p.z = pack2(v1[0], v1[1]); p.w = pack2(v1[2], v1[3]);
        *(uint4*)(ap + j) = p;
      }
    } else {
      int src = eidx[e];
      int dst = eidx[Etot + e];
      DSTS[le] = dst;
      const float* es = e_s + (size_t)e * 8;
      for (int j = 0; j < 8; ++j) ap[128 + j] = f2bf(es[j]);
      const float* sp = S + (size_t)dst * 128;
      for (int j = 0; j < 128; j += 8) {
        f32x4u v0 = *(const f32x4u*)(sp + j);
        f32x4u v1 = *(const f32x4u*)(sp + j + 4);
        uint4 p;
        p.x = pack2(v0[0], v0[1]); p.y = pack2(v0[2], v0[3]);
        p.z = pack2(v1[0], v1[1]); p.w = pack2(v1[2], v1[3]);
        *(uint4*)(ap + 136 + j) = p;
      }
      float mv[5][3];
      const float* vs = x_v + (size_t)src * 6;
      const float* vd = x_v + (size_t)dst * 6;
      const float* ev = e_v + (size_t)e * 3;
      #pragma unroll
      for (int c = 0; c < 3; ++c) {
        mv[0][c] = vs[c]; mv[1][c] = vs[3+c];
        mv[2][c] = ev[c];
        mv[3][c] = vd[c]; mv[4][c] = vd[3+c];
      }
      #pragma unroll
      for (int h = 0; h < 5; ++h) {
        float a0 = 0.f, a1 = 0.f, a2 = 0.f;
        #pragma unroll
        for (int vch = 0; vch < 5; ++vch) {
          float wgt = wh0[h*5+vch];
          a0 += wgt*mv[vch][0]; a1 += wgt*mv[vch][1]; a2 += wgt*mv[vch][2];
        }
        VH0[le][h*3+0]=a0; VH0[le][h*3+1]=a1; VH0[le][h*3+2]=a2;
        ap[264+h] = f2bf(sqrtf(fmaxf(a0*a0+a1*a1+a2*a2, EPSN)));
      }
      for (int j = 269; j < 288; ++j) ap[j] = 0;
    }
  }
  __syncthreads();

  int arow = wave * 16 + (lane & 15);
  int aoff = (lane >> 4) << 3;
  int koff = (lane >> 4) << 3;
  int bnr  = lane & 15;

  // ---- GVP0 scalar GEMM: [64x288]x[288x128]
  v8f acc[8] = {};
  for (int kt = 0; kt < 288; kt += 32) {
    const unsigned short* ab = A + arow * A0S + kt + aoff;
    uint4 alo = *(const uint4*)(ab);
    uint4 ahi = *(const uint4*)(ab + 16);
    const unsigned short* bb = B0T + (size_t)bnr * 288 + kt + koff;
    uint4 blo[8], bhi[8];
    #pragma unroll
    for (int t = 0; t < 8; ++t) {
      blo[t] = *(const uint4*)(bb + (size_t)t*16*288);
      bhi[t] = *(const uint4*)(bb + (size_t)t*16*288 + 16);
    }
    #pragma unroll
    for (int t = 0; t < 8; ++t)
      acc[t] = wmma_bf16(alo, ahi, blo[t], bhi[t], acc[t]);
  }
  __syncthreads();
  { // epilogue0: relu(.+bs0) -> A1 cols 0..127
    int lr0 = wave * 16 + ((lane >> 4) << 3);
    int col = lane & 15;
    #pragma unroll
    for (int t = 0; t < 8; ++t) {
      int c = t*16 + col; float b = bs0[c];
      #pragma unroll
      for (int r = 0; r < 8; ++r)
        A[(lr0 + r) * A1S + c] = f2bf(fmaxf(acc[t][r] + b, 0.f));
    }
  }
  if (tid < 64) { // vector stage 1: gate(wv0 x vh0) -> vh1, norms -> A1[128..129]
    int le = tid;
    float vo[2][3];
    #pragma unroll
    for (int o = 0; o < 2; ++o) {
      float a0 = 0.f, a1 = 0.f, a2 = 0.f;
      #pragma unroll
      for (int h = 0; h < 5; ++h) {
        float wgt = wv0[o*5+h];
        a0 += wgt*VH0[le][h*3]; a1 += wgt*VH0[le][h*3+1]; a2 += wgt*VH0[le][h*3+2];
      }
      float g = sigmoidf(sqrtf(fmaxf(a0*a0+a1*a1+a2*a2, EPSN)));
      vo[o][0]=a0*g; vo[o][1]=a1*g; vo[o][2]=a2*g;
    }
    unsigned short* ap = A + le * A1S;
    #pragma unroll
    for (int h = 0; h < 2; ++h) {
      float a0 = wh1[h*2]*vo[0][0] + wh1[h*2+1]*vo[1][0];
      float a1 = wh1[h*2]*vo[0][1] + wh1[h*2+1]*vo[1][1];
      float a2 = wh1[h*2]*vo[0][2] + wh1[h*2+1]*vo[1][2];
      VH[le][h*3]=a0; VH[le][h*3+1]=a1; VH[le][h*3+2]=a2;
      ap[128+h] = f2bf(sqrtf(fmaxf(a0*a0+a1*a1+a2*a2, EPSN)));
    }
    for (int j = 130; j < 160; ++j) ap[j] = 0;
  }
  __syncthreads();

  // ---- GVP1 scalar GEMM: [64x160]x[160x128]
  v8f acc1[8] = {};
  for (int kt = 0; kt < 160; kt += 32) {
    const unsigned short* ab = A + arow * A1S + kt + aoff;
    uint4 alo = *(const uint4*)(ab);
    uint4 ahi = *(const uint4*)(ab + 16);
    const unsigned short* bb = B1T + (size_t)bnr * 160 + kt + koff;
    uint4 blo[8], bhi[8];
    #pragma unroll
    for (int t = 0; t < 8; ++t) {
      blo[t] = *(const uint4*)(bb + (size_t)t*16*160);
      bhi[t] = *(const uint4*)(bb + (size_t)t*16*160 + 16);
    }
    #pragma unroll
    for (int t = 0; t < 8; ++t)
      acc1[t] = wmma_bf16(alo, ahi, blo[t], bhi[t], acc1[t]);
  }
  __syncthreads();
  { // epilogue1: relu(.+bs1) -> A2 cols 0..127
    int lr0 = wave * 16 + ((lane >> 4) << 3);
    int col = lane & 15;
    #pragma unroll
    for (int t = 0; t < 8; ++t) {
      int c = t*16 + col; float b = bs1[c];
      #pragma unroll
      for (int r = 0; r < 8; ++r)
        A[(lr0 + r) * A1S + c] = f2bf(fmaxf(acc1[t][r] + b, 0.f));
    }
  }
  if (tid < 64) { // vector stage 2: gate(wv1 x vh1) -> vh2
    int le = tid;
    float vo[2][3];
    #pragma unroll
    for (int o = 0; o < 2; ++o) {
      float a0 = wv1[o*2]*VH[le][0] + wv1[o*2+1]*VH[le][3];
      float a1 = wv1[o*2]*VH[le][1] + wv1[o*2+1]*VH[le][4];
      float a2 = wv1[o*2]*VH[le][2] + wv1[o*2+1]*VH[le][5];
      float g = sigmoidf(sqrtf(fmaxf(a0*a0+a1*a1+a2*a2, EPSN)));
      vo[o][0]=a0*g; vo[o][1]=a1*g; vo[o][2]=a2*g;
    }
    float nv[6];
    #pragma unroll
    for (int h = 0; h < 2; ++h) {
      nv[h*3+0] = wh2[h*2]*vo[0][0] + wh2[h*2+1]*vo[1][0];
      nv[h*3+1] = wh2[h*2]*vo[0][1] + wh2[h*2+1]*vo[1][1];
      nv[h*3+2] = wh2[h*2]*vo[0][2] + wh2[h*2+1]*vo[1][2];
    }
    unsigned short* ap = A + le * A1S;
    ap[128] = f2bf(sqrtf(fmaxf(nv[0]*nv[0]+nv[1]*nv[1]+nv[2]*nv[2], EPSN)));
    ap[129] = f2bf(sqrtf(fmaxf(nv[3]*nv[3]+nv[4]*nv[4]+nv[5]*nv[5], EPSN)));
    #pragma unroll
    for (int j = 0; j < 6; ++j) VH[le][j] = nv[j];
  }
  __syncthreads();

  // ---- GVP2 scalar GEMM: [64x160]x[160x128], no activation
  v8f acc2[8] = {};
  for (int kt = 0; kt < 160; kt += 32) {
    const unsigned short* ab = A + arow * A1S + kt + aoff;
    uint4 alo = *(const uint4*)(ab);
    uint4 ahi = *(const uint4*)(ab + 16);
    const unsigned short* bb = B2T + (size_t)bnr * 160 + kt + koff;
    uint4 blo[8], bhi[8];
    #pragma unroll
    for (int t = 0; t < 8; ++t) {
      blo[t] = *(const uint4*)(bb + (size_t)t*16*160);
      bhi[t] = *(const uint4*)(bb + (size_t)t*16*160 + 16);
    }
    #pragma unroll
    for (int t = 0; t < 8; ++t)
      acc2[t] = wmma_bf16(alo, ahi, blo[t], bhi[t], acc2[t]);
  }
  { // scatter scalar messages (mean aggregation numerator)
    int lr0 = wave * 16 + ((lane >> 4) << 3);
    int col = lane & 15;
    #pragma unroll
    for (int t = 0; t < 8; ++t) {
      int c = t*16 + col; float b = bs2[c];
      #pragma unroll
      for (int r = 0; r < 8; ++r) {
        if (e0 + lr0 + r < Etot) {
          int dst = DSTS[lr0 + r];
          atomicAdd(&aggS[(size_t)dst * 128 + c], acc2[t][r] + b);
        }
      }
    }
  }
  if (tid < 64 && e0 + tid < Etot) { // scatter vector messages + degree count
    int le = tid; int dst = DSTS[le];
    #pragma unroll
    for (int o = 0; o < 2; ++o) {
      float a0 = wv2[o*2]*VH[le][0] + wv2[o*2+1]*VH[le][3];
      float a1 = wv2[o*2]*VH[le][1] + wv2[o*2+1]*VH[le][4];
      float a2 = wv2[o*2]*VH[le][2] + wv2[o*2+1]*VH[le][5];
      atomicAdd(&aggV[(size_t)dst*6 + o*3 + 0], a0);
      atomicAdd(&aggV[(size_t)dst*6 + o*3 + 1], a1);
      atomicAdd(&aggV[(size_t)dst*6 + o*3 + 2], a2);
    }
    atomicAdd(&cnt[dst], 1.0f);
  }
}

// ---------------------------------------------------------------- residual + LN0 + FF input prep
__global__ void __launch_bounds__(256)
k_node(int Nn, const float* __restrict__ S, const float* __restrict__ aggS,
       const float* __restrict__ aggV, const float* __restrict__ cnt,
       const float* __restrict__ x_v,
       const float* __restrict__ ln0g, const float* __restrict__ ln0b,
       const float* __restrict__ whf0, const float* __restrict__ wvf0,
       const float* __restrict__ whf1,
       float* __restrict__ SLN, unsigned short* __restrict__ AF0,
       unsigned short* __restrict__ AF1)
{
  int wave = threadIdx.x >> 5, lane = threadIdx.x & 31;
  int n = blockIdx.x * 8 + wave;
  if (n >= Nn) return;
  float rin = 1.f / fmaxf(cnt[n], 1.f);
  float v[4]; float sum = 0.f, sq = 0.f;
  #pragma unroll
  for (int j = 0; j < 4; ++j) {
    int c = lane + j * 32;
    float x = S[(size_t)n*128 + c] + aggS[(size_t)n*128 + c] * rin;
    v[j] = x; sum += x; sq += x*x;
  }
  #pragma unroll
  for (int m = 1; m < 32; m <<= 1) { sum += __shfl_xor(sum, m, 32); sq += __shfl_xor(sq, m, 32); }
  float mu = sum * (1.f/128.f);
  float inv = rsqrtf(sq * (1.f/128.f) - mu*mu + 1e-5f);
  #pragma unroll
  for (int j = 0; j < 4; ++j) {
    int c = lane + j * 32;
    float y = (v[j] - mu) * inv * ln0g[c] + ln0b[c];
    SLN[(size_t)n*128 + c] = y;
    AF0[(size_t)n*160 + c] = f2bf(y);
  }
  if (lane == 0) {
    float vl[2][3]; float nrm[2];
    #pragma unroll
    for (int o = 0; o < 2; ++o) {
      float d = 0.f;
      #pragma unroll
      for (int c = 0; c < 3; ++c) {
        float x = x_v[(size_t)n*6 + o*3 + c] + aggV[(size_t)n*6 + o*3 + c] * rin;
        vl[o][c] = x; d += x*x;
      }
      nrm[o] = fmaxf(d, EPSN);
    }
    float ivn = rsqrtf(0.5f * (nrm[0] + nrm[1]));
    #pragma unroll
    for (int o = 0; o < 2; ++o)
      #pragma unroll
      for (int c = 0; c < 3; ++c) vl[o][c] *= ivn;
    float vh[4][3];
    unsigned short* a0 = AF0 + (size_t)n*160;
    #pragma unroll
    for (int h = 0; h < 4; ++h) {
      float d = 0.f;
      #pragma unroll
      for (int c = 0; c < 3; ++c) {
        float x = whf0[h*2]*vl[0][c] + whf0[h*2+1]*vl[1][c];
        vh[h][c] = x; d += x*x;
      }
      a0[128+h] = f2bf(sqrtf(fmaxf(d, EPSN)));
    }
    for (int j = 132; j < 160; ++j) a0[j] = 0;
    float vo[4][3];
    #pragma unroll
    for (int o = 0; o < 4; ++o) {
      float a[3] = {0.f,0.f,0.f};
      #pragma unroll
      for (int h = 0; h < 4; ++h) {
        float wgt = wvf0[o*4+h];
        a[0] += wgt*vh[h][0]; a[1] += wgt*vh[h][1]; a[2] += wgt*vh[h][2];
      }
      float g = sigmoidf(sqrtf(fmaxf(a[0]*a[0]+a[1]*a[1]+a[2]*a[2], EPSN)));
      vo[o][0]=a[0]*g; vo[o][1]=a[1]*g; vo[o][2]=a[2]*g;
    }
    unsigned short* a1 = AF1 + (size_t)n*544;
    #pragma unroll
    for (int h = 0; h < 4; ++h) {
      float a[3] = {0.f,0.f,0.f};
      #pragma unroll
      for (int o = 0; o < 4; ++o) {
        float wgt = whf1[h*4+o];
        a[0] += wgt*vo[o][0]; a[1] += wgt*vo[o][1]; a[2] += wgt*vo[o][2];
      }
      a1[512+h] = f2bf(sqrtf(fmaxf(a[0]*a[0]+a[1]*a[1]+a[2]*a[2], EPSN)));
    }
    for (int j = 516; j < 544; ++j) a1[j] = 0;
  }
}

// ---------------------------------------------------------------- FF0: [M,160]x[160,512], relu
__global__ void __launch_bounds__(512)
k_ff0(int M, const unsigned short* __restrict__ Amat, const unsigned short* __restrict__ BT,
      const float* __restrict__ bias, unsigned short* __restrict__ Aout)
{
  int tid = threadIdx.x, wave = tid >> 5, lane = tid & 31;
  int m0 = blockIdx.x * 64;
  int rowg = wave & 3, colg = wave >> 2;        // 16 waves: 4 row groups x 4 col groups (128 cols)
  v8f acc[8] = {};
  int arow = m0 + rowg*16 + (lane & 15); if (arow >= M) arow = M - 1;
  int aoff = (lane >> 4) << 3;
  int koff = (lane >> 4) << 3;
  for (int kt = 0; kt < 160; kt += 32) {
    uint4 alo = *(const uint4*)(Amat + (size_t)arow*160 + kt + aoff);
    uint4 ahi = *(const uint4*)(Amat + (size_t)arow*160 + kt + 16 + aoff);
    const unsigned short* bb = BT + (size_t)(colg*128 + (lane & 15)) * 160 + kt + koff;
    uint4 blo[8], bhi[8];
    #pragma unroll
    for (int t = 0; t < 8; ++t) {
      blo[t] = *(const uint4*)(bb + (size_t)t*16*160);
      bhi[t] = *(const uint4*)(bb + (size_t)t*16*160 + 16);
    }
    #pragma unroll
    for (int t = 0; t < 8; ++t)
      acc[t] = wmma_bf16(alo, ahi, blo[t], bhi[t], acc[t]);
  }
  int r0 = m0 + rowg*16 + ((lane >> 4) << 3);
  int col0 = colg*128 + (lane & 15);
  #pragma unroll
  for (int t = 0; t < 8; ++t) {
    int c = col0 + t*16; float b = bias[c];
    #pragma unroll
    for (int r = 0; r < 8; ++r) {
      int row = r0 + r;
      if (row < M) Aout[(size_t)row*544 + c] = f2bf(fmaxf(acc[t][r] + b, 0.f));
    }
  }
}

// ---------------------------------------------------------------- FF1 + LN1 + readout sigmoid
__global__ void __launch_bounds__(256)
k_ff1(int M, const unsigned short* __restrict__ Amat, const unsigned short* __restrict__ BT,
      const float* __restrict__ bias, const float* __restrict__ SLN,
      const float* __restrict__ g1, const float* __restrict__ b1,
      const float* __restrict__ row_w, const float* __restrict__ row_b,
      float* __restrict__ out)
{
  __shared__ float DS[64 * 132];
  int tid = threadIdx.x, wave = tid >> 5, lane = tid & 31;
  int m0 = blockIdx.x * 64;
  int rowg = wave & 3, colg = wave >> 2;      // colg 0..1
  v8f acc[4] = {};
  int arow = m0 + rowg*16 + (lane & 15); if (arow >= M) arow = M - 1;
  int aoff = (lane >> 4) << 3;
  int koff = (lane >> 4) << 3;
  for (int kt = 0; kt < 544; kt += 32) {
    uint4 alo = *(const uint4*)(Amat + (size_t)arow*544 + kt + aoff);
    uint4 ahi = *(const uint4*)(Amat + (size_t)arow*544 + kt + 16 + aoff);
    const unsigned short* bb = BT + (size_t)(colg*64 + (lane & 15)) * 544 + kt + koff;
    uint4 blo[4], bhi[4];
    #pragma unroll
    for (int t = 0; t < 4; ++t) {
      blo[t] = *(const uint4*)(bb + (size_t)t*16*544);
      bhi[t] = *(const uint4*)(bb + (size_t)t*16*544 + 16);
    }
    #pragma unroll
    for (int t = 0; t < 4; ++t)
      acc[t] = wmma_bf16(alo, ahi, blo[t], bhi[t], acc[t]);
  }
  int lr0 = rowg*16 + ((lane >> 4) << 3);
  int col0 = colg*64 + (lane & 15);
  #pragma unroll
  for (int t = 0; t < 4; ++t) {
    int c = col0 + t*16; float b = bias[c];
    #pragma unroll
    for (int r = 0; r < 8; ++r) DS[(lr0 + r) * 132 + c] = acc[t][r] + b;
  }
  __syncthreads();
  int lr = tid >> 2, q = tid & 3;             // 4 threads per row, 32 cols each
  int row = m0 + lr;
  int rc = row < M ? row : M - 1;
  float vals[32]; float sum = 0.f, sq = 0.f;
  for (int j = 0; j < 32; ++j) {
    int c = q*32 + j;
    float x = SLN[(size_t)rc*128 + c] + DS[lr*132 + c];
    vals[j] = x; sum += x; sq += x*x;
  }
  sum += __shfl_xor(sum, 1, 32); sum += __shfl_xor(sum, 2, 32);
  sq  += __shfl_xor(sq, 1, 32);  sq  += __shfl_xor(sq, 2, 32);
  float mu = sum * (1.f/128.f);
  float inv = rsqrtf(sq * (1.f/128.f) - mu*mu + 1e-5f);
  float dot = 0.f;
  for (int j = 0; j < 32; ++j) {
    int c = q*32 + j;
    float y = (vals[j] - mu) * inv * g1[c] + b1[c];
    dot += y * row_w[c];
  }
  dot += __shfl_xor(dot, 1, 32); dot += __shfl_xor(dot, 2, 32);
  if (q == 0 && row < M) out[row] = sigmoidf(dot + row_b[0]);
}

// ---------------------------------------------------------------- host
extern "C" void kernel_launch(void* const* d_in, const int* in_sizes, int n_in,
                              void* d_out, int out_size, void* d_ws, size_t ws_size,
                              hipStream_t stream)
{
  (void)ws_size;
  const void* in[33];
  if (n_in >= 33) {
    // map insertion-order slot -> index in alphabetically-sorted leaf order
    static const int smap[33] = {9,8,15,16,14,17,19,20,18,21,23,24,22,25,
                                 1,2,0,3,5,6,4,7,11,10,13,12,27,26,28,29,30,31,32};
    bool sorted = (in_sizes[0] == 512);
    for (int i = 0; i < 33; ++i) in[i] = d_in[sorted ? smap[i] : i];
  } else {
    // params passed as one flat buffer (insertion order)
    static const int lsz[28] = {183424,128,25,34432,128,10,4,16640,128,4,4,16640,128,4,
                                8,67584,512,16,16,66048,128,8,128,128,128,128,128,1};
    const float* base = (const float*)d_in[0];
    size_t o = 0;
    for (int i = 0; i < 28; ++i) { in[i] = (const void*)(base + o); o += lsz[i]; }
    for (int i = 0; i < 5; ++i) in[28 + i] = d_in[n_in - 5 + i];
  }
  const float* lin_w=(const float*)in[0];  const float* lin_b=(const float*)in[1];
  const float* wh0=(const float*)in[2];    const float* ws0=(const float*)in[3];
  const float* bs0=(const float*)in[4];    const float* wv0=(const float*)in[5];
  const float* wh1=(const float*)in[6];    const float* ws1=(const float*)in[7];
  const float* bs1=(const float*)in[8];    const float* wv1=(const float*)in[9];
  const float* wh2=(const float*)in[10];   const float* ws2=(const float*)in[11];
  const float* bs2=(const float*)in[12];   const float* wv2=(const float*)in[13];
  const float* whf0=(const float*)in[14];  const float* wsf0=(const float*)in[15];
  const float* bsf0=(const float*)in[16];  const float* wvf0=(const float*)in[17];
  const float* whf1=(const float*)in[18];  const float* wsf1=(const float*)in[19];
  const float* bsf1=(const float*)in[20];  /* wvf1 (in[21]) never affects the output */
  const float* ln0g=(const float*)in[22];  const float* ln0b=(const float*)in[23];
  const float* ln1g=(const float*)in[24];  const float* ln1b=(const float*)in[25];
  const float* ro_w=(const float*)in[26];  const float* ro_b=(const float*)in[27];
  const float* x_s=(const float*)in[28];   const float* x_v=(const float*)in[29];
  const int*   eidx=(const int*)in[30];
  const float* e_s=(const float*)in[31];   const float* e_v=(const float*)in[32];

  int Nn = out_size;                                   // 20000
  int Etot = (n_in >= 33) ? in_sizes[31] / 8 : in_sizes[n_in - 2] / 8;  // 320000

  char* wsp = (char*)d_ws;
  size_t off = 0;
  auto give = [&](size_t bytes) -> char* {
    char* p = wsp + off; off = (off + bytes + 255) & ~(size_t)255; return p;
  };
  unsigned short* BlinT = (unsigned short*)give((size_t)128*1440*2);
  unsigned short* B0T   = (unsigned short*)give((size_t)128*288*2);
  unsigned short* B1T   = (unsigned short*)give((size_t)128*160*2);
  unsigned short* B2T   = (unsigned short*)give((size_t)128*160*2);
  unsigned short* Bf0T  = (unsigned short*)give((size_t)512*160*2);
  unsigned short* Bf1T  = (unsigned short*)give((size_t)128*544*2);
  float* S    = (float*)give((size_t)Nn*128*4);
  float* AGGS = (float*)give((size_t)Nn*128*4);
  float* AGGV = (float*)give((size_t)Nn*6*4);
  float* CNT  = (float*)give((size_t)Nn*4);
  float* SLN  = (float*)give((size_t)Nn*128*4);
  unsigned short* AF0 = (unsigned short*)give((size_t)Nn*160*2);
  unsigned short* AF1 = (unsigned short*)give((size_t)Nn*544*2);

  k_zero<<<(Nn*128+255)/256, 256, 0, stream>>>(AGGS, Nn*128);
  k_zero<<<(Nn*6+255)/256,  256, 0, stream>>>(AGGV, Nn*6);
  k_zero<<<(Nn+255)/256,    256, 0, stream>>>(CNT, Nn);

  k_convert_t<<<(128*1440+255)/256, 256, 0, stream>>>(lin_w, BlinT, 1433, 128, 1440);
  k_convert_t<<<(128*288+255)/256,  256, 0, stream>>>(ws0,  B0T, 269, 128, 288);
  k_convert_t<<<(128*160+255)/256,  256, 0, stream>>>(ws1,  B1T, 130, 128, 160);
  k_convert_t<<<(128*160+255)/256,  256, 0, stream>>>(ws2,  B2T, 130, 128, 160);
  k_convert_t<<<(512*160+255)/256,  256, 0, stream>>>(wsf0, Bf0T, 132, 512, 160);
  k_convert_t<<<(128*544+255)/256,  256, 0, stream>>>(wsf1, Bf1T, 516, 128, 544);

  k_gemm_lin<<<(Nn+127)/128, 256, 0, stream>>>(x_s, BlinT, lin_b, S, Nn, 1433);
  k_edge<<<(Etot+63)/64, 128, 0, stream>>>(Etot, eidx, e_s, e_v, x_v, S,
      B0T, B1T, B2T, bs0, bs1, bs2, wh0, wv0, wh1, wv1, wh2, wv2, AGGS, AGGV, CNT);
  k_node<<<(Nn+7)/8, 256, 0, stream>>>(Nn, S, AGGS, AGGV, CNT, x_v, ln0g, ln0b,
      whf0, wvf0, whf1, SLN, AF0, AF1);
  k_ff0<<<(Nn+63)/64, 512, 0, stream>>>(Nn, AF0, Bf0T, bsf0, AF1);
  k_ff1<<<(Nn+63)/64, 256, 0, stream>>>(Nn, AF1, Bf1T, bsf1, SLN, ln1g, ln1b,
      ro_w, ro_b, (float*)d_out);
}
